// MultiTaskCGCNN_678604832948
// MI455X (gfx1250) — matrix operand order repaired and checked by reading, not compiled, not run on hardware
//
#include <hip/hip_runtime.h>
#include <hip/hip_bf16.h>
#include <math.h>

// ---------------------------------------------------------------------------
// MultiTaskCGCNN on MI455X (gfx1250, wave32, WMMA f32<-f16 16x16x32)
//
//   z @ W  =  h[dst] @ W[0:64]  +  h[src] @ W[64:128]  +  e @ W[128:169]
//
//   P[n, 0:64]   = h[n] @ Wf[0:64]      P[n, 64:128] = h[n] @ Wf[64:128]
//   P[n,128:192] = h[n] @ Ws[0:64]      P[n,192:256] = h[n] @ Ws[64:128]
//
// Edge kernel computes e @ W[128:169] with WMMA (K padded 41->64) from
// LDS-staged pre-swizzled B fragments, adds gathered P parts + bias,
// applies sigmoid*softplus, atomic-scatters into agg.
// P + h + agg ~ 154MB -> L2 (192MB); HBM streams only edge_attr.
// ---------------------------------------------------------------------------

#define NN      100000
#define NE      1600000
#define NGRAPH  256
#define NODEF   92
#define EDGEF   41
#define HIDN    64

typedef __attribute__((ext_vector_type(16))) _Float16 v16h;
typedef __attribute__((ext_vector_type(8)))  float    v8f;

#define WMMA_F32_F16(A, B, C) \
    __builtin_amdgcn_wmma_f32_16x16x32_f16(false, (A), false, (B), (short)0, (C), false, false)

// ---- A-matrix 16x32 f16 fragment from row-major f32 (ISA 7.12.2 layout) ----
// lane<16: M=lane, K={kb+0..7, kb+16..23}; lane>=16: same M, K={kb+8..15, kb+24..31}
// Requires 16B-aligned (row*ld + k0)*4 for the vector path (ld=64 or 92: OK).
__device__ __forceinline__ v16h load_a_f32_vec(const float* __restrict__ S, int ld,
                                               int row, int kb, int kmax, int lane) {
    int r  = row + (lane & 15);
    int hh = (lane >> 4) & 1;
    const float* p = S + (long long)r * ld;
    v16h a;
    int k0 = kb + hh * 8;
    int k1 = kb + 16 + hh * 8;
    if (k0 + 7 < kmax) {
        float4 q0 = *reinterpret_cast<const float4*>(p + k0);
        float4 q1 = *reinterpret_cast<const float4*>(p + k0 + 4);
        a[0] = (_Float16)q0.x; a[1] = (_Float16)q0.y; a[2] = (_Float16)q0.z; a[3] = (_Float16)q0.w;
        a[4] = (_Float16)q1.x; a[5] = (_Float16)q1.y; a[6] = (_Float16)q1.z; a[7] = (_Float16)q1.w;
    } else {
#pragma unroll
        for (int i = 0; i < 8; ++i)
            a[i] = (_Float16)((k0 + i < kmax) ? p[k0 + i] : 0.0f);
    }
    if (k1 + 7 < kmax) {
        float4 q0 = *reinterpret_cast<const float4*>(p + k1);
        float4 q1 = *reinterpret_cast<const float4*>(p + k1 + 4);
        a[8]  = (_Float16)q0.x; a[9]  = (_Float16)q0.y; a[10] = (_Float16)q0.z; a[11] = (_Float16)q0.w;
        a[12] = (_Float16)q1.x; a[13] = (_Float16)q1.y; a[14] = (_Float16)q1.z; a[15] = (_Float16)q1.w;
    } else {
#pragma unroll
        for (int i = 0; i < 8; ++i)
            a[8 + i] = (_Float16)((k1 + i < kmax) ? p[k1 + i] : 0.0f);
    }
    return a;
}

// ---- pre-swizzled B fragments in LDS: frag f, lane l -> 16 f16 = 32B -------
// element i of lane l corresponds to  K = kb + (l>>4)*8 + (i&7) + (i>>3)*16,  N = col0 + (l&15)
__device__ __forceinline__ v16h read_frag(const _Float16* buf, int f, int lane) {
    return *reinterpret_cast<const v16h*>(buf + f * 512 + lane * 16);
}

// ---------------------------------------------------------------------------
// K1: h = relu(x @ emb_W + emb_b)    [100000,92] x [92,64], K padded 92->96
// 12 B fragments (4 N-tiles x 3 K-steps) packed once per block into LDS.
// ---------------------------------------------------------------------------
__global__ void cgcnn_embed_kernel(const float* __restrict__ x,
                                   const float* __restrict__ W,
                                   const float* __restrict__ bias,
                                   float* __restrict__ hout, int nN) {
    __shared__ __align__(32) _Float16 frags[12 * 512];   // 12 KB
    for (int idx = threadIdx.x; idx < 12 * 512; idx += blockDim.x) {
        int f = idx >> 9, l = (idx >> 4) & 31, i = idx & 15;
        int t = f / 3, ks = f % 3;
        int k = ks * 32 + ((l >> 4) & 1) * 8 + (i & 7) + ((i >> 3) & 1) * 16;
        int n = t * 16 + (l & 15);
        frags[idx] = (_Float16)((k < NODEF) ? W[(long long)k * HIDN + n] : 0.0f);
    }
    __syncthreads();

    int lane = threadIdx.x & 31;
    int wave = threadIdx.x >> 5;
    int tile = blockIdx.x * (blockDim.x >> 5) + wave;
    if (tile * 16 >= nN) return;
    int row = tile * 16;

    v16h a0 = load_a_f32_vec(x, NODEF, row, 0,  NODEF, lane);
    v16h a1 = load_a_f32_vec(x, NODEF, row, 32, NODEF, lane);
    v16h a2 = load_a_f32_vec(x, NODEF, row, 64, NODEF, lane);

    int hh = (lane >> 4) & 1, nlo = lane & 15;
#pragma unroll
    for (int t = 0; t < 4; ++t) {
        v8f acc = {};
        acc = WMMA_F32_F16(a0, read_frag(frags, t * 3 + 0, lane), acc);
        acc = WMMA_F32_F16(a1, read_frag(frags, t * 3 + 1, lane), acc);
        acc = WMMA_F32_F16(a2, read_frag(frags, t * 3 + 2, lane), acc);
        float bb = bias[t * 16 + nlo];
#pragma unroll
        for (int v = 0; v < 8; ++v) {
            int r = row + v + 8 * hh;
            hout[(long long)r * HIDN + t * 16 + nlo] = fmaxf(acc[v] + bb, 0.0f);
        }
    }
}

// ---------------------------------------------------------------------------
// K2: P = h @ [Wf[0:64] | Wf[64:128] | Ws[0:64] | Ws[64:128]]   [N,64]x[64,256]
// 32 B fragments (16 col-tiles x 2 K-steps) packed once per block into LDS.
// ---------------------------------------------------------------------------
__global__ void cgcnn_nodeproj_kernel(const float* __restrict__ h,
                                      const float* __restrict__ Wf,
                                      const float* __restrict__ Ws,
                                      float* __restrict__ P, int nN) {
    __shared__ __align__(32) _Float16 frags[32 * 512];   // 32 KB
    for (int idx = threadIdx.x; idx < 32 * 512; idx += blockDim.x) {
        int f = idx >> 9, l = (idx >> 4) & 31, i = idx & 15;
        int t = f >> 1, ks = f & 1;
        int blk = t >> 2;                                // 0:Wf-dst 1:Wf-src 2:Ws-dst 3:Ws-src
        const float* W = ((blk < 2) ? Wf : Ws) + (long long)((blk & 1) * 64) * HIDN;
        int k = ks * 32 + ((l >> 4) & 1) * 8 + (i & 7) + ((i >> 3) & 1) * 16;       // 0..63, always valid
        int n = (t & 3) * 16 + (l & 15);
        frags[idx] = (_Float16)W[(long long)k * HIDN + n];
    }
    __syncthreads();

    int lane = threadIdx.x & 31;
    int wave = threadIdx.x >> 5;
    int tile = blockIdx.x * (blockDim.x >> 5) + wave;
    if (tile * 16 >= nN) return;
    int row = tile * 16;

    v16h a0 = load_a_f32_vec(h, HIDN, row, 0,  HIDN, lane);
    v16h a1 = load_a_f32_vec(h, HIDN, row, 32, HIDN, lane);

    int hh = (lane >> 4) & 1, nlo = lane & 15;
#pragma unroll
    for (int t = 0; t < 16; ++t) {
        v8f acc = {};
        acc = WMMA_F32_F16(a0, read_frag(frags, t * 2 + 0, lane), acc);
        acc = WMMA_F32_F16(a1, read_frag(frags, t * 2 + 1, lane), acc);
#pragma unroll
        for (int v = 0; v < 8; ++v) {
            int r = row + v + 8 * hh;
            P[(long long)r * 256 + t * 16 + nlo] = acc[v];
        }
    }
}

// ---------------------------------------------------------------------------
// K3: fused edge kernel.  One wave per 16-edge tile:
//   accF/accS = edge_attr_tile @ W{f,s}[128:169]  (2 K-steps x 4 N-tiles WMMA)
//   pre = acc + P[dst,part] + P[src,part] + bias ; msg = sigmoid*softplus
//   atomicAdd into agg[dst]
// 16 B fragments (2 mats x 2 K-steps x 4 N-tiles) packed once per block.
// ---------------------------------------------------------------------------
__global__ void cgcnn_edge_kernel(const float* __restrict__ P,
                                  const float* __restrict__ edge_attr,
                                  const int*   __restrict__ ei,   // [2, E]
                                  const float* __restrict__ Wf,   // [169,64]
                                  const float* __restrict__ Ws,
                                  const float* __restrict__ bf,
                                  const float* __restrict__ bs,
                                  float* __restrict__ agg, int nE) {
    __shared__ __align__(32) _Float16 frags[16 * 512];   // 16 KB
    for (int idx = threadIdx.x; idx < 16 * 512; idx += blockDim.x) {
        int f = idx >> 9, l = (idx >> 4) & 31, i = idx & 15;
        int mat = f >> 3, ks = (f >> 2) & 1, t = f & 3;
        const float* W = mat ? Ws : Wf;
        int k = ks * 32 + ((l >> 4) & 1) * 8 + (i & 7) + ((i >> 3) & 1) * 16;
        int n = t * 16 + (l & 15);
        frags[idx] = (_Float16)((k < EDGEF) ? W[(long long)(128 + k) * HIDN + n] : 0.0f);
    }
    __syncthreads();

    int lane = threadIdx.x & 31;
    int wave = threadIdx.x >> 5;
    int tile = blockIdx.x * (blockDim.x >> 5) + wave;
    if (tile * 16 >= nE) return;           // wave-uniform: EXEC stays full
    int ebase = tile * 16;

    // A fragments: 16 edges x K(0..63, valid 0..40) from edge_attr [E,41]
    // (row stride 164B is not 16B-aligned -> element loads; this is the
    //  streamed-from-HBM operand, byte volume is unchanged)
    v16h a0, a1;
    {
        int r  = ebase + (lane & 15);
        int hf = (lane >> 4) & 1;
        const float* p = edge_attr + (long long)r * EDGEF;
#pragma unroll
        for (int i = 0; i < 8; ++i) {
            int k0 = hf * 8 + i;           // 0..15   < 41
            int k1 = 16 + hf * 8 + i;      // 16..31  < 41
            int k2 = 32 + hf * 8 + i;      // 32..47  (pad >= 41)
            a0[i]     = (_Float16)p[k0];
            a0[i + 8] = (_Float16)p[k1];
            a1[i]     = (_Float16)((k2 < EDGEF) ? p[k2] : 0.0f);
            a1[i + 8] = (_Float16)0.0f;    // K 48..63 pad
        }
    }

    int hh = (lane >> 4) & 1, nlo = lane & 15;
    long long dsts[8], srcs[8];
#pragma unroll
    for (int v = 0; v < 8; ++v) {
        int e   = ebase + v + 8 * hh;      // edge row of acc VGPR v for this lane
        srcs[v] = (long long)ei[e];        // x_j
        dsts[v] = (long long)ei[nE + e];   // x_i
    }

#pragma unroll
    for (int t = 0; t < 4; ++t) {
        int col = t * 16 + nlo;
        v8f accF = {}, accS = {};
        accF = WMMA_F32_F16(a0, read_frag(frags, 0 * 8 + 0 * 4 + t, lane), accF);
        accF = WMMA_F32_F16(a1, read_frag(frags, 0 * 8 + 1 * 4 + t, lane), accF);
        accS = WMMA_F32_F16(a0, read_frag(frags, 1 * 8 + 0 * 4 + t, lane), accS);
        accS = WMMA_F32_F16(a1, read_frag(frags, 1 * 8 + 1 * 4 + t, lane), accS);

        float bfc = bf[col], bsc = bs[col];
#pragma unroll
        for (int v = 0; v < 8; ++v) {
            long long dd = dsts[v], ss = srcs[v];
            float pf = accF[v] + P[dd * 256 +       col] + P[ss * 256 +  64 + col] + bfc;
            float ps = accS[v] + P[dd * 256 + 128 + col] + P[ss * 256 + 192 + col] + bsc;
            float gate = 1.0f / (1.0f + __expf(-pf));
            float sp   = (ps > 20.0f) ? ps : __logf(1.0f + __expf(ps));
            atomicAdd(&agg[dd * HIDN + col], gate * sp);
        }
    }
}

// ---------------------------------------------------------------------------
// K4: h = relu(h + agg)   (float4-vectorized)
// ---------------------------------------------------------------------------
__global__ void cgcnn_resrelu_kernel(float4* __restrict__ h,
                                     const float4* __restrict__ agg, int n4) {
    int i = blockIdx.x * blockDim.x + threadIdx.x;
    if (i >= n4) return;
    float4 a = h[i], b = agg[i];
    a.x = fmaxf(a.x + b.x, 0.0f);
    a.y = fmaxf(a.y + b.y, 0.0f);
    a.z = fmaxf(a.z + b.z, 0.0f);
    a.w = fmaxf(a.w + b.w, 0.0f);
    h[i] = a;
}

// ---------------------------------------------------------------------------
// K5: mean-pool accumulation exploiting sorted batch ids: each thread owns
// one column of a 16-node run and emits one atomic per segment change.
// ---------------------------------------------------------------------------
__global__ void cgcnn_pool_kernel(const float* __restrict__ h,
                                  const int* __restrict__ batch,
                                  float* __restrict__ sums,
                                  float* __restrict__ cnt, int nN) {
    int c        = threadIdx.x & 63;
    int chunk    = (blockIdx.x * (blockDim.x >> 6)) + (threadIdx.x >> 6);
    long long n0 = (long long)chunk * 16;
    if (n0 >= nN) return;
    long long n1 = (n0 + 16 < (long long)nN) ? n0 + 16 : (long long)nN;

    int   curg = batch[n0];
    float acc = 0.0f, ccnt = 0.0f;
    for (long long n = n0; n < n1; ++n) {
        int g = batch[n];
        if (g != curg) {
            atomicAdd(&sums[(long long)curg * HIDN + c], acc);
            if (c == 0) atomicAdd(&cnt[curg], ccnt);
            curg = g; acc = 0.0f; ccnt = 0.0f;
        }
        acc += h[n * HIDN + c];
        ccnt += 1.0f;
    }
    atomicAdd(&sums[(long long)curg * HIDN + c], acc);
    if (c == 0) atomicAdd(&cnt[curg], ccnt);
}

// ---------------------------------------------------------------------------
// K6: head: out = relu(pooled @ l1_W + l1_b) @ l2_W + l2_b    (tiny)
// ---------------------------------------------------------------------------
__global__ void cgcnn_head_kernel(const float* __restrict__ sums,
                                  const float* __restrict__ cnt,
                                  const float* __restrict__ W1,
                                  const float* __restrict__ b1,
                                  const float* __restrict__ W2,
                                  const float* __restrict__ b2,
                                  float* __restrict__ out) {
    int g = blockIdx.x * blockDim.x + threadIdx.x;
    if (g >= NGRAPH) return;
    float inv = 1.0f / fmaxf(cnt[g], 1.0f);
    float pooled[HIDN];
#pragma unroll
    for (int i = 0; i < HIDN; ++i) pooled[i] = sums[(long long)g * HIDN + i] * inv;
    float hid[32];
#pragma unroll 4
    for (int j = 0; j < 32; ++j) {
        float s = b1[j];
#pragma unroll
        for (int i = 0; i < HIDN; ++i) s += pooled[i] * W1[i * 32 + j];
        hid[j] = fmaxf(s, 0.0f);
    }
#pragma unroll
    for (int k = 0; k < 2; ++k) {
        float s = b2[k];
#pragma unroll
        for (int j = 0; j < 32; ++j) s += hid[j] * W2[j * 2 + k];
        out[g * 2 + k] = s;
    }
}

// ---------------------------------------------------------------------------
extern "C" void kernel_launch(void* const* d_in, const int* in_sizes, int n_in,
                              void* d_out, int out_size, void* d_ws, size_t ws_size,
                              hipStream_t stream) {
    const float* x      = (const float*)d_in[0];
    const int*   ei     = (const int*)  d_in[1];
    const float* ea     = (const float*)d_in[2];
    const int*   batch  = (const int*)  d_in[3];
    const float* embW   = (const float*)d_in[4];
    const float* embB   = (const float*)d_in[5];
    const float* cWf[2] = { (const float*)d_in[6],  (const float*)d_in[10] };
    const float* cbf[2] = { (const float*)d_in[7],  (const float*)d_in[11] };
    const float* cWs[2] = { (const float*)d_in[8],  (const float*)d_in[12] };
    const float* cbs[2] = { (const float*)d_in[9],  (const float*)d_in[13] };
    const float* l1W    = (const float*)d_in[14];
    const float* l1b    = (const float*)d_in[15];
    const float* l2W    = (const float*)d_in[16];
    const float* l2b    = (const float*)d_in[17];
    float* out = (float*)d_out;

    // workspace layout (f32): h | agg | P | sums | cnt   (~154 MB -> L2-resident)
    float* h    = (float*)d_ws;
    float* agg  = h   + (size_t)NN * HIDN;
    float* P    = agg + (size_t)NN * HIDN;
    float* sums = P   + (size_t)NN * 256;
    float* cnt  = sums + (size_t)NGRAPH * HIDN;

    const int wavesPerBlock = 8;
    const dim3 blk(32 * wavesPerBlock);

    const int nodeTiles  = NN / 16;                               // 6250
    const int nodeBlocks = (nodeTiles + wavesPerBlock - 1) / wavesPerBlock;
    const int edgeTiles  = NE / 16;                               // 100000
    const int edgeBlocks = (edgeTiles + wavesPerBlock - 1) / wavesPerBlock;

    // 1) embedding
    cgcnn_embed_kernel<<<nodeBlocks, blk, 0, stream>>>(x, embW, embB, h, NN);

    // 2) two CGConv layers
    for (int layer = 0; layer < 2; ++layer) {
        cgcnn_nodeproj_kernel<<<nodeBlocks, blk, 0, stream>>>(h, cWf[layer], cWs[layer], P, NN);
        hipMemsetAsync(agg, 0, (size_t)NN * HIDN * sizeof(float), stream);
        cgcnn_edge_kernel<<<edgeBlocks, blk, 0, stream>>>(P, ea, ei, cWf[layer], cWs[layer],
                                                          cbf[layer], cbs[layer], agg, NE);
        cgcnn_resrelu_kernel<<<(NN * HIDN / 4 + 255) / 256, 256, 0, stream>>>(
            (float4*)h, (const float4*)agg, NN * HIDN / 4);
    }

    // 3) mean pool + head
    hipMemsetAsync(sums, 0, (size_t)(NGRAPH * HIDN + NGRAPH) * sizeof(float), stream);
    {
        int chunks = (NN + 15) / 16;                              // 6250
        int blocks = (chunks + 3) / 4;                            // 4 chunks / 256-thread block
        cgcnn_pool_kernel<<<blocks, 256, 0, stream>>>(h, batch, sums, cnt, NN);
    }
    cgcnn_head_kernel<<<1, NGRAPH, 0, stream>>>(sums, cnt, l1W, l1b, l2W, l2b, out);
}
